// TransformerBlock_9878424781410
// MI455X (gfx1250) — compile-verified, matrix operand order checked by reading, not run on hardware
//
#include <hip/hip_runtime.h>
#include <cmath>

// ---------------------------------------------------------------------------
// Transformer block for MI455X (gfx1250), bf16 WMMA pipeline, fp32 accum.
// B=2, S=2048, D=1024, H=16, d=64, FF=4096.
// GEMMs: double-buffered async global->LDS staging (ASYNCcnt, counted waits)
//        + v_wmma_f32_16x16x32_bf16.
// Attention: flash-style, base-2 softmax, WMMA row sums, ds_swizzle row max.
// ---------------------------------------------------------------------------

typedef __bf16 bf16_t;
typedef __attribute__((ext_vector_type(16))) __bf16 v16bf;
typedef __attribute__((ext_vector_type(8)))  float  v8f;

#define EMB   1024
#define SEQ   2048
#define BATCH 2
#define HEADS 16
#define HDIM  64
#define FFDIM 4096
#define TOKENS (BATCH * SEQ)   // 4096

// 0.125 (1/sqrt(64)) / ln(2): folded into Q so softmax runs in base 2.
#define QSCALE 0.1803368801111204f

static __device__ __forceinline__ v8f wmma_bf16(v16bf a, v16bf b, v8f c) {
    return __builtin_amdgcn_wmma_f32_16x16x32_bf16(false, a, false, b,
                                                   (short)0, c, false, false);
}

// CDNA5 16-bit A/B fragment: elements 0..7 -> K = k0+8*half+e,
// elements 8..15 -> K = k0+16+8*half+(e-8).  Two 16B vector loads.
static __device__ __forceinline__ v16bf load_frag(const bf16_t* lo, const bf16_t* hi) {
    v16bf r;
    ((uint4*)&r)[0] = *(const uint4*)lo;
    ((uint4*)&r)[1] = *(const uint4*)hi;
    return r;
}

// Async global -> LDS 16-byte copy (ASYNCcnt). Generic LDS address low 32
// bits == LDS byte address (LDS aperture rule).
static __device__ __forceinline__ void async_copy16(void* lds_dst, const void* gsrc) {
    const unsigned lds_off = (unsigned)(unsigned long long)lds_dst;
    asm volatile("global_load_async_to_lds_b128 %0, %1, off"
                 :: "v"(lds_off), "v"(gsrc) : "memory");
}
static __device__ __forceinline__ void wait_async0() {
    asm volatile("s_wait_asynccnt 0x0" ::: "memory");
}
// Allow the 6 just-issued copies of the NEXT tile to remain in flight;
// in-order async completion => the previous tile's 6 copies have landed.
static __device__ __forceinline__ void wait_async6() {
    asm volatile("s_wait_asynccnt 0x6" ::: "memory");
}
static __device__ __forceinline__ void wait_ds0() {
    asm volatile("s_wait_dscnt 0x0" ::: "memory");
}

// Max over the 16-lane C-layout group via ds_swizzle XOR patterns.
static __device__ __forceinline__ float row_max16(float x) {
    x = fmaxf(x, __int_as_float(__builtin_amdgcn_ds_swizzle(__float_as_int(x), 0x041F)));
    x = fmaxf(x, __int_as_float(__builtin_amdgcn_ds_swizzle(__float_as_int(x), 0x081F)));
    x = fmaxf(x, __int_as_float(__builtin_amdgcn_ds_swizzle(__float_as_int(x), 0x101F)));
    x = fmaxf(x, __int_as_float(__builtin_amdgcn_ds_swizzle(__float_as_int(x), 0x201F)));
    return x;
}

// ---------------------------------------------------------------------------
// Weight transpose + fp32 -> bf16:  Wt[n*K + k] = W[k*N + n]
// ---------------------------------------------------------------------------
__global__ void k_transpose_bf16(const float* __restrict__ W,
                                 bf16_t* __restrict__ Wt, int K, int N) {
    int idx = blockIdx.x * blockDim.x + threadIdx.x;
    if (idx >= K * N) return;
    int k = idx / N;
    int n = idx - k * N;
    Wt[(size_t)n * K + k] = (bf16_t)W[idx];
}

// ---------------------------------------------------------------------------
// LayerNorm (unbiased variance, ddof=1) fp32 -> bf16, one block (256t) per row.
// ---------------------------------------------------------------------------
__global__ __launch_bounds__(256)
void k_layernorm_bf16(const float* __restrict__ X,
                      const float* __restrict__ g,
                      const float* __restrict__ s,
                      bf16_t* __restrict__ H) {
    __shared__ float rsum[256];
    __shared__ float rsq[256];
    const int row = blockIdx.x;
    const int t   = threadIdx.x;

    const float4 v  = ((const float4*)(X + (size_t)row * EMB))[t];
    rsum[t] = v.x + v.y + v.z + v.w;
    rsq[t]  = v.x * v.x + v.y * v.y + v.z * v.z + v.w * v.w;
    __syncthreads();
    for (int off = 128; off > 0; off >>= 1) {
        if (t < off) { rsum[t] += rsum[t + off]; rsq[t] += rsq[t + off]; }
        __syncthreads();
    }
    const float mean = rsum[0] * (1.0f / EMB);
    const float var  = (rsq[0] - (float)EMB * mean * mean) * (1.0f / (EMB - 1));
    const float inv  = rsqrtf(var + 1e-5f);

    const float4 gg = ((const float4*)g)[t];
    const float4 ss = ((const float4*)s)[t];
    bf16_t* h = H + (size_t)row * EMB + t * 4;
    h[0] = (bf16_t)((v.x - mean) * inv * gg.x + ss.x);
    h[1] = (bf16_t)((v.y - mean) * inv * gg.y + ss.y);
    h[2] = (bf16_t)((v.z - mean) * inv * gg.z + ss.z);
    h[3] = (bf16_t)((v.w - mean) * inv * gg.w + ss.w);
}

// ---------------------------------------------------------------------------
// GEMM: C[M,N] = A[M,K] * Bt[N,K]^T   (A,Bt bf16, fp32 accum)
// 256 threads = 8 waves; block tile 128x64, wave tile 16x64, k-step 64.
// Double-buffered async global->LDS staging overlaps DMA with WMMA.
// ---------------------------------------------------------------------------
enum GemmMode {
    MODE_HEADS   = 0,  // scatter bf16*oscale -> [B,H,S,d]     (Q, K projections)
    MODE_HEADS_T = 1,  // scatter bf16*oscale -> [B,H,d,S]     (V projection)
    MODE_BIASRES = 2,  // fp32 out = acc + bias[n] + res[m,n]  (Wo, FFN2)
    MODE_FF1     = 3   // bf16 out = gelu(acc + bias[n])       (FFN1)
};

static __device__ __forceinline__ void stage_tiles(
    const bf16_t* __restrict__ A, const bf16_t* __restrict__ Bt,
    int K, int mrow0, int ntile, int tid, int k0,
    bf16_t* bufA, bf16_t* bufB) {
    // A tile 128x64 bf16 = 16 KB: 4 chunks of 16B per thread (128B per row).
#pragma unroll
    for (int c = 0; c < 4; ++c) {
        const int l   = (tid + c * 256) * 16;
        const int row = l >> 7;
        const char* src = (const char*)(A + (size_t)(mrow0 + row) * K + k0) + (l & 127);
        async_copy16((char*)bufA + l, src);
    }
    // B tile 64x64 bf16 = 8 KB: 2 chunks per thread.
#pragma unroll
    for (int c = 0; c < 2; ++c) {
        const int l   = (tid + c * 256) * 16;
        const int row = l >> 7;
        const char* src = (const char*)(Bt + (size_t)(ntile + row) * K + k0) + (l & 127);
        async_copy16((char*)bufB + l, src);
    }
}

template <int MODE>
__global__ __launch_bounds__(256)
void k_gemm(const bf16_t* __restrict__ A,
            const bf16_t* __restrict__ Bt,
            const float* __restrict__ bias,
            const float* __restrict__ res,
            void* __restrict__ out,
            int M, int N, int K, float oscale) {
    __shared__ __align__(16) bf16_t lds_a[2][128 * 64];   // 2 x 16 KB
    __shared__ __align__(16) bf16_t lds_b[2][64 * 64];    // 2 x 8 KB

    const int tid   = threadIdx.x;
    const int lane  = tid & 31;
    const int wave  = tid >> 5;                   // 0..7
    const int half  = lane >> 4;                  // 0/1
    const int lr    = lane & 15;                  // row (A) / col (B,C)
    const int mtile = blockIdx.y * 128 + wave * 16;
    const int ntile = blockIdx.x * 64;
    const int mrow0 = blockIdx.y * 128;

    v8f acc[4];
#pragma unroll
    for (int j = 0; j < 4; ++j)
#pragma unroll
        for (int r = 0; r < 8; ++r) acc[j][r] = 0.0f;

    const int NK = K >> 6;                        // 64-wide k tiles
    stage_tiles(A, Bt, K, mrow0, ntile, tid, 0, lds_a[0], lds_b[0]);

    for (int kt = 0; kt < NK; ++kt) {
        const int cur = kt & 1;
        if (kt + 1 < NK) {
            stage_tiles(A, Bt, K, mrow0, ntile, tid, (kt + 1) << 6,
                        lds_a[cur ^ 1], lds_b[cur ^ 1]);
            wait_async6();          // tile kt landed; kt+1 still in flight
        } else {
            wait_async0();
        }
        __syncthreads();

        const bf16_t* abase = lds_a[cur] + (wave * 16 + lr) * 64;
#pragma unroll
        for (int kk = 0; kk < 64; kk += 32) {
            const v16bf af = load_frag(abase + kk + 8 * half,
                                       abase + kk + 16 + 8 * half);
#pragma unroll
            for (int j = 0; j < 4; ++j) {
                const bf16_t* bcol = lds_b[cur] + (j * 16 + lr) * 64 + kk;
                const v16bf bf = load_frag(bcol + 8 * half, bcol + 16 + 8 * half);
                acc[j] = wmma_bf16(af, bf, acc[j]);
            }
        }
        __syncthreads();            // all waves done reading before overwrite
    }

    // Epilogue. C layout: lane (half, lr) + vgpr r -> (m = r + 8*half, n = lr).
#pragma unroll
    for (int j = 0; j < 4; ++j) {
#pragma unroll
        for (int r = 0; r < 8; ++r) {
            const int m = mtile + r + 8 * half;
            const int n = ntile + j * 16 + lr;
            float v = acc[j][r];
            if (MODE == MODE_HEADS || MODE == MODE_HEADS_T) {
                const int b  = m >> 11;           // SEQ = 2048
                const int si = m & (SEQ - 1);
                const int h  = n >> 6;            // HDIM = 64
                const int dd = n & (HDIM - 1);
                bf16_t* o = (bf16_t*)out;
                const size_t bh = (size_t)(b * HEADS + h);
                if (MODE == MODE_HEADS)
                    o[(bh * SEQ + si) * HDIM + dd] = (bf16_t)(v * oscale);
                else
                    o[(bh * HDIM + dd) * SEQ + si] = (bf16_t)(v * oscale);
            } else if (MODE == MODE_BIASRES) {
                float* o = (float*)out;
                o[(size_t)m * N + n] = v + bias[n] + res[(size_t)m * N + n];
            } else { // MODE_FF1: bias + tanh-GELU -> bf16
                const float tv = v + bias[n];
                const float u  = 0.7978845608028654f * (tv + 0.044715f * tv * tv * tv);
                const float gel = 0.5f * tv * (1.0f + tanhf(u));
                ((bf16_t*)out)[(size_t)m * N + n] = (bf16_t)gel;
            }
        }
    }
}

// ---------------------------------------------------------------------------
// Causal flash attention: one wave per (b*h, 16-query block).
// Q pre-scaled by 0.125/ln2 -> softmax in base 2 (v_exp_f32 is exp2).
// Row sums via P x ones WMMA; row max via ds_swizzle XOR reduction.
// ---------------------------------------------------------------------------
template <bool MASKED>
static __device__ __forceinline__ void attn_tile(
    int kb, int q0, int half, int lr,
    const bf16_t* __restrict__ kbase, const bf16_t* __restrict__ vbase,
    const v16bf& qf0, const v16bf& qf1, const v16bf& ones,
    bf16_t* ptile, v8f (&acc)[4], float (&mprev)[8], float (&lsum)[8]) {

    // ---- scores (already in log2 units): two 16x16 tiles ----
    v8f s0, s1;
#pragma unroll
    for (int r = 0; r < 8; ++r) { s0[r] = 0.0f; s1[r] = 0.0f; }

    const bf16_t* kc0 = kbase + (size_t)(kb + lr) * HDIM;
    const bf16_t* kc1 = kbase + (size_t)(kb + 16 + lr) * HDIM;
    {
        v16bf kf = load_frag(kc0 + 0  + 8 * half, kc0 + 16 + 8 * half);
        s0 = wmma_bf16(qf0, kf, s0);
        kf = load_frag(kc0 + 32 + 8 * half, kc0 + 48 + 8 * half);
        s0 = wmma_bf16(qf1, kf, s0);
        kf = load_frag(kc1 + 0  + 8 * half, kc1 + 16 + 8 * half);
        s1 = wmma_bf16(qf0, kf, s1);
        kf = load_frag(kc1 + 32 + 8 * half, kc1 + 48 + 8 * half);
        s1 = wmma_bf16(qf1, kf, s1);
    }

    // ---- online base-2 softmax ----
#pragma unroll
    for (int r = 0; r < 8; ++r) {
        const int m = r + 8 * half;                 // query row in tile
        float v0 = s0[r];
        float v1 = s1[r];
        if (MASKED) {
            if (kb + lr      > q0 + m) v0 = -3.0e38f;
            if (kb + 16 + lr > q0 + m) v1 = -3.0e38f;
        }
        const float tm   = row_max16(fmaxf(v0, v1));
        const float mnew = fmaxf(mprev[r], tm);
        const float corr = exp2f(mprev[r] - mnew);
        mprev[r] = mnew;
        lsum[r] *= corr;
#pragma unroll
        for (int j = 0; j < 4; ++j) acc[j][r] *= corr;

        ptile[m * 32 + lr]      = (bf16_t)exp2f(v0 - mnew);
        ptile[m * 32 + 16 + lr] = (bf16_t)exp2f(v1 - mnew);
    }

    // LDS store -> load relayout (C layout -> A-frag layout)
    wait_ds0();
    const v16bf pf = load_frag(&ptile[lr * 32 + 8 * half],
                               &ptile[lr * 32 + 16 + 8 * half]);

    // ---- row sums via P x ones WMMA: D[m][*] = sum_k P[m][k] ----
    {
        v8f rs;
#pragma unroll
        for (int r = 0; r < 8; ++r) rs[r] = 0.0f;
        rs = wmma_bf16(pf, ones, rs);
#pragma unroll
        for (int r = 0; r < 8; ++r) lsum[r] += rs[r];
    }

    // ---- ctx += P(16x32) * V(32x64): V columns contiguous via VT ----
#pragma unroll
    for (int j = 0; j < 4; ++j) {
        const bf16_t* vcol = vbase + (size_t)(j * 16 + lr) * SEQ + kb;
        const v16bf vf = load_frag(vcol + 8 * half, vcol + 16 + 8 * half);
        acc[j] = wmma_bf16(pf, vf, acc[j]);
    }
    wait_ds0();   // guard ptile reuse next iteration
}

__global__ __launch_bounds__(32)
void k_attention(const bf16_t* __restrict__ Q,
                 const bf16_t* __restrict__ Kmat,
                 const bf16_t* __restrict__ VT,
                 bf16_t* __restrict__ CTX) {
    __shared__ __align__(16) bf16_t ptile[16 * 32];   // P tile relayout buffer

    const int qb   = blockIdx.x;          // 0..S/16-1
    const int bh   = blockIdx.y;          // 0..B*H-1
    const int lane = threadIdx.x & 31;
    const int half = lane >> 4;
    const int lr   = lane & 15;
    const int q0   = qb * 16;

    const bf16_t* qbase = Q    + (size_t)bh * SEQ * HDIM;
    const bf16_t* kbase = Kmat + (size_t)bh * SEQ * HDIM;
    const bf16_t* vbase = VT   + (size_t)bh * HDIM * SEQ;

    const bf16_t* qrow = qbase + (size_t)(q0 + lr) * HDIM;
    const v16bf qf0 = load_frag(qrow + 0  + 8 * half, qrow + 16 + 8 * half);
    const v16bf qf1 = load_frag(qrow + 32 + 8 * half, qrow + 48 + 8 * half);

    v16bf ones;
#pragma unroll
    for (int e = 0; e < 16; ++e) ones[e] = (bf16_t)1.0f;

    v8f acc[4];
    float mprev[8], lsum[8];
#pragma unroll
    for (int j = 0; j < 4; ++j)
#pragma unroll
        for (int r = 0; r < 8; ++r) acc[j][r] = 0.0f;
#pragma unroll
    for (int r = 0; r < 8; ++r) { mprev[r] = -3.0e38f; lsum[r] = 0.0f; }

    // Causal: keys 0 .. q0+15; only the LAST 32-key tile touches the diagonal.
    const int kend  = q0 + 16;
    const int klast = ((kend - 1) >> 5) << 5;
    for (int kb = 0; kb < klast; kb += 32)
        attn_tile<false>(kb, q0, half, lr, kbase, vbase, qf0, qf1, ones,
                         ptile, acc, mprev, lsum);
    attn_tile<true>(klast, q0, half, lr, kbase, vbase, qf0, qf1, ones,
                    ptile, acc, mprev, lsum);

    // ---- normalize + scatter ctx to [token, h*64 + d] row-major bf16 ----
    const int b = bh >> 4, h = bh & (HEADS - 1);
#pragma unroll
    for (int j = 0; j < 4; ++j) {
#pragma unroll
        for (int r = 0; r < 8; ++r) {
            const int m   = r + 8 * half;
            const int tok = b * SEQ + q0 + m;
            const int n   = h * HDIM + j * 16 + lr;
            CTX[(size_t)tok * EMB + n] = (bf16_t)(acc[j][r] / lsum[r]);
        }
    }
}

// ---------------------------------------------------------------------------
// Host-side orchestration
// ---------------------------------------------------------------------------
extern "C" void kernel_launch(void* const* d_in, const int* in_sizes, int n_in,
                              void* d_out, int out_size, void* d_ws, size_t ws_size,
                              hipStream_t stream) {
    (void)in_sizes; (void)n_in; (void)out_size; (void)ws_size;

    const float* x  = (const float*)d_in[0];
    const float* Wq = (const float*)d_in[1];
    const float* Wk = (const float*)d_in[2];
    const float* Wv = (const float*)d_in[3];
    const float* Wo = (const float*)d_in[4];
    const float* bo = (const float*)d_in[5];
    const float* W1 = (const float*)d_in[6];
    const float* b1 = (const float*)d_in[7];
    const float* W2 = (const float*)d_in[8];
    const float* b2 = (const float*)d_in[9];
    const float* g1 = (const float*)d_in[10];
    const float* s1 = (const float*)d_in[11];
    const float* g2 = (const float*)d_in[12];
    const float* s2 = (const float*)d_in[13];

    // Workspace layout (bf16 elements, all MB-scale aligned)
    bf16_t* p    = (bf16_t*)d_ws;
    bf16_t* WqT  = p; p += (size_t)EMB * EMB;
    bf16_t* WkT  = p; p += (size_t)EMB * EMB;
    bf16_t* WvT  = p; p += (size_t)EMB * EMB;
    bf16_t* WoT  = p; p += (size_t)EMB * EMB;
    bf16_t* W1T  = p; p += (size_t)EMB * FFDIM;   // [FF][EMB]
    bf16_t* W2T  = p; p += (size_t)FFDIM * EMB;   // [EMB][FF]
    bf16_t* H1   = p; p += (size_t)TOKENS * EMB;
    bf16_t* Qb   = p; p += (size_t)TOKENS * EMB;  // [B,H,S,d] (pre-scaled)
    bf16_t* Kb   = p; p += (size_t)TOKENS * EMB;  // [B,H,S,d]
    bf16_t* VTb  = p; p += (size_t)TOKENS * EMB;  // [B,H,d,S]
    bf16_t* CTX  = p; p += (size_t)TOKENS * EMB;  // [tok, EMB]
    bf16_t* H2   = p; p += (size_t)TOKENS * EMB;
    bf16_t* A1   = p; p += (size_t)TOKENS * FFDIM;
    float*  X1   = (float*)p;                     // [tok, EMB] fp32

    const dim3 t256(256);

    // 1) weights -> bf16, transposed to [N][K]
    {
        int n = EMB * EMB;
        k_transpose_bf16<<<(n + 255) / 256, t256, 0, stream>>>(Wq, WqT, EMB, EMB);
        k_transpose_bf16<<<(n + 255) / 256, t256, 0, stream>>>(Wk, WkT, EMB, EMB);
        k_transpose_bf16<<<(n + 255) / 256, t256, 0, stream>>>(Wv, WvT, EMB, EMB);
        k_transpose_bf16<<<(n + 255) / 256, t256, 0, stream>>>(Wo, WoT, EMB, EMB);
        n = EMB * FFDIM;
        k_transpose_bf16<<<(n + 255) / 256, t256, 0, stream>>>(W1, W1T, EMB, FFDIM);
        k_transpose_bf16<<<(n + 255) / 256, t256, 0, stream>>>(W2, W2T, FFDIM, EMB);
    }

    // 2) LN1
    k_layernorm_bf16<<<TOKENS, t256, 0, stream>>>(x, g1, s1, H1);

    // 3) QKV projections (Q pre-scaled by 0.125/ln2 for base-2 softmax)
    const dim3 gD(EMB / 64, TOKENS / 128);        // (16, 32)
    k_gemm<MODE_HEADS><<<gD, t256, 0, stream>>>(H1, WqT, nullptr, nullptr, Qb,
                                                TOKENS, EMB, EMB, QSCALE);
    k_gemm<MODE_HEADS><<<gD, t256, 0, stream>>>(H1, WkT, nullptr, nullptr, Kb,
                                                TOKENS, EMB, EMB, 1.0f);
    k_gemm<MODE_HEADS_T><<<gD, t256, 0, stream>>>(H1, WvT, nullptr, nullptr, VTb,
                                                  TOKENS, EMB, EMB, 1.0f);

    // 4) causal flash attention
    k_attention<<<dim3(SEQ / 16, BATCH * HEADS), 32, 0, stream>>>(Qb, Kb, VTb, CTX);

    // 5) output projection + residual -> X1 (fp32)
    k_gemm<MODE_BIASRES><<<gD, t256, 0, stream>>>(CTX, WoT, bo, x, X1,
                                                  TOKENS, EMB, EMB, 1.0f);

    // 6) LN2
    k_layernorm_bf16<<<TOKENS, t256, 0, stream>>>(X1, g2, s2, H2);

    // 7) FFN1 + GELU -> A1 (bf16)
    const dim3 gF(FFDIM / 64, TOKENS / 128);      // (64, 32)
    k_gemm<MODE_FF1><<<gF, t256, 0, stream>>>(H2, W1T, b1, nullptr, A1,
                                              TOKENS, FFDIM, EMB, 1.0f);

    // 8) FFN2 + residual -> d_out (fp32)
    k_gemm<MODE_BIASRES><<<gD, t256, 0, stream>>>(A1, W2T, b2, X1, (float*)d_out,
                                                  TOKENS, EMB, FFDIM, 1.0f);
}
